// Enformer_70222715290284
// MI455X (gfx1250) — matrix operand order repaired and checked by reading, not compiled, not run on hardware
//
#include <hip/hip_runtime.h>
#include <hip/hip_bf16.h>

typedef __attribute__((ext_vector_type(16))) _Float16 v16h;
typedef __attribute__((ext_vector_type(8)))  _Float16 v8h;
typedef __attribute__((ext_vector_type(8)))  float    v8f;

// LDS row stride in halves: 144 bytes -> 16B alignment for every b128 DS access.
#define LDSH 72

// gfx1250 async global->LDS staging (ASYNCcnt-tracked DMA, no VGPR round-trip).
#if defined(__AMDGCN__) && defined(__has_builtin)
#if __has_builtin(__builtin_amdgcn_global_load_async_to_lds_b128) && \
    __has_builtin(__builtin_amdgcn_s_wait_asynccnt)
#define ENF_ASYNC_LDS 1
#endif
#endif

#ifdef ENF_ASYNC_LDS
// builtin expects: (v4i AS1*, v4i AS3*, imm offset, imm cpol)
typedef int enf_v4i __attribute__((vector_size(16)));
typedef enf_v4i __attribute__((address_space(1)))* enf_gp;
typedef enf_v4i __attribute__((address_space(3)))* enf_lp;
#endif

enum { EPI_STORE = 0, EPI_BIAS = 1, EPI_SCATTER = 2, EPI_STORE16 = 3 };

// ---------------------------------------------------------------------------
// WMMA GEMM: C[M,N] = A[M,K] * Bt[N,K]^T   (B always given pre-transposed f16)
// Block = 256 threads = 8 waves; block tile 128x128, BK = 64.
// Wave w: wm = w>>1 (32-row strip, 2 M-tiles), wn = w&1 (64-col strip, 4 N-tiles)
// -> 8 accumulators (16x16 f32) per wave, 16 WMMAs per outer k-step.
// AT = float (cvt to f16 at LDS-store) or _Float16 (async b128 DMA to LDS).
// Per-z offsets:
//   A += z*azs ; B += (z/bzdiv)*bs_hi + (z%bzdiv)*bs_lo
//   C += (z/czdiv)*cs_hi + (z%czdiv)*cs_lo   (element units of the C dtype)
// EPI_SCATTER: relative shift, logits[m, m + n - (nctx-1)] += acc.
// EPI_STORE16: store f16 (C reinterpreted as _Float16*).
// ---------------------------------------------------------------------------
template<typename AT, int EPI>
__global__ __launch_bounds__(256)
void enf_gemm_wmma(const AT* __restrict__ A, const _Float16* __restrict__ B,
                   float* __restrict__ C, const float* __restrict__ bias,
                   int M, int N, int K,
                   int lda, int ldb, int ldc,
                   long azs,
                   long bs_hi, long bs_lo, int bzdiv,
                   long cs_hi, long cs_lo, int czdiv,
                   int nctx)
{
  __shared__ _Float16 As[128 * LDSH];
  __shared__ _Float16 Bs[128 * LDSH];

  const int z = blockIdx.z;
  const AT*       Ab = A + (long)z * azs;
  const _Float16* Bb = B + (long)(z / bzdiv) * bs_hi + (long)(z % bzdiv) * bs_lo;
  const long zoffC   = (long)(z / czdiv) * cs_hi + (long)(z % czdiv) * cs_lo;

  const int row0 = blockIdx.y * 128;
  const int col0 = blockIdx.x * 128;
  const int tid  = threadIdx.x;
  const int lane = tid & 31;
  const int wave = tid >> 5;
  const int wm   = wave >> 1;   // 0..3 : rows wm*32 .. +32
  const int wn   = wave & 1;    // 0..1 : cols wn*64 .. +64

  v8f acc[2][4];
  #pragma unroll
  for (int mi = 0; mi < 2; ++mi)
    #pragma unroll
    for (int ni = 0; ni < 4; ++ni) acc[mi][ni] = (v8f){};

  // staging: each thread copies 32 contiguous K-elements of one row
  const int sr = tid >> 1;          // 0..127
  const int sk = (tid & 1) * 32;    // 0 or 32
  int arow = row0 + sr; if (arow >= M) arow = M - 1;
  int brow = col0 + sr; if (brow >= N) brow = N - 1;

  const int hiA = (lane >> 4) * 8;   // A frag: K {0..7,16..23} / {8..15,24..31}
  const int hiB = (lane >> 4) * 8;   // B frag: K {0..7,16..23} / {8..15,24..31}
  const int l15 = lane & 15;

  for (int k0 = 0; k0 < K; k0 += 64) {
    // ---- stage A tile (128 x 64) as [m][k] ----
    {
      const AT* src = Ab + (long)arow * lda + (k0 + sk);
      _Float16* dst = &As[sr * LDSH + sk];
      if (sizeof(AT) == 2) {
#ifdef ENF_ASYNC_LDS
        __builtin_amdgcn_global_load_async_to_lds_b128((enf_gp)src, (enf_lp)dst, 0, 0);
        __builtin_amdgcn_global_load_async_to_lds_b128((enf_gp)src, (enf_lp)dst, 16, 0);
        __builtin_amdgcn_global_load_async_to_lds_b128((enf_gp)src, (enf_lp)dst, 32, 0);
        __builtin_amdgcn_global_load_async_to_lds_b128((enf_gp)src, (enf_lp)dst, 48, 0);
#else
        const v8h* s = (const v8h*)(const void*)src;
        v8h* d = (v8h*)dst;
        #pragma unroll
        for (int e = 0; e < 4; ++e) d[e] = s[e];
#endif
      } else {
        #pragma unroll
        for (int e = 0; e < 32; ++e) dst[e] = (_Float16)src[e];
      }
    }
    // ---- stage B tile (128 x 64) as [n][k] (already transposed in memory) ----
    {
      const _Float16* src = Bb + (long)brow * ldb + (k0 + sk);
      _Float16* dst = &Bs[sr * LDSH + sk];
#ifdef ENF_ASYNC_LDS
      __builtin_amdgcn_global_load_async_to_lds_b128((enf_gp)src, (enf_lp)dst, 0, 0);
      __builtin_amdgcn_global_load_async_to_lds_b128((enf_gp)src, (enf_lp)dst, 16, 0);
      __builtin_amdgcn_global_load_async_to_lds_b128((enf_gp)src, (enf_lp)dst, 32, 0);
      __builtin_amdgcn_global_load_async_to_lds_b128((enf_gp)src, (enf_lp)dst, 48, 0);
#else
      const v8h* s = (const v8h*)src;
      v8h* d = (v8h*)dst;
      #pragma unroll
      for (int e = 0; e < 4; ++e) d[e] = s[e];
#endif
    }
    if (k0 + 64 < K) {  // gfx1250 global_prefetch_b8
      __builtin_prefetch(Ab + (long)arow * lda + (k0 + 64 + sk), 0, 1);
      __builtin_prefetch(Bb + (long)brow * ldb + (k0 + 64 + sk), 0, 1);
    }
#ifdef ENF_ASYNC_LDS
    __builtin_amdgcn_s_wait_asynccnt(0);   // async DMA done before barrier
#endif
    __syncthreads();

    #pragma unroll
    for (int ks = 0; ks < 2; ++ks) {
      const int kb = ks * 32;
      // pre-load all fragments of this k-substep so ds_loads can be claused
      v16h afr[2];
      #pragma unroll
      for (int mi = 0; mi < 2; ++mi) {
        const int mloc = wm * 32 + mi * 16 + l15;
        v8h alo = *(const v8h*)&As[mloc * LDSH + kb + hiA];
        v8h ahi = *(const v8h*)&As[mloc * LDSH + kb + 16 + hiA];
        afr[mi] = __builtin_shufflevector(alo, ahi,
            0,1,2,3,4,5,6,7,8,9,10,11,12,13,14,15);
      }
      v16h bfr[4];
      #pragma unroll
      for (int ni = 0; ni < 4; ++ni) {
        const int nloc = wn * 64 + ni * 16 + l15;
        v8h b0 = *(const v8h*)&Bs[nloc * LDSH + kb + hiB];
        v8h b1 = *(const v8h*)&Bs[nloc * LDSH + kb + hiB + 16];
        bfr[ni] = __builtin_shufflevector(b0, b1,
            0,1,2,3,4,5,6,7,8,9,10,11,12,13,14,15);
      }
      #pragma unroll
      for (int ni = 0; ni < 4; ++ni) {
        acc[0][ni] = __builtin_amdgcn_wmma_f32_16x16x32_f16(
            false, afr[0], false, bfr[ni], (short)0, acc[0][ni], false, false);
        acc[1][ni] = __builtin_amdgcn_wmma_f32_16x16x32_f16(
            false, afr[1], false, bfr[ni], (short)0, acc[1][ni], false, false);
      }
    }
    __syncthreads();
  }

  // ---- epilogue: D layout VGPR r -> M = r + 8*(lane>=16), N = lane%16 ----
  const int dm = (lane >> 4) * 8;
  #pragma unroll
  for (int mi = 0; mi < 2; ++mi) {
    #pragma unroll
    for (int ni = 0; ni < 4; ++ni) {
      #pragma unroll
      for (int r = 0; r < 8; ++r) {
        const int mrow = row0 + wm * 32 + mi * 16 + dm + r;
        const int ncol = col0 + wn * 64 + ni * 16 + l15;
        const float v = acc[mi][ni][r];
        if (EPI == EPI_STORE) {
          if (mrow < M && ncol < N) C[zoffC + (long)mrow * ldc + ncol] = v;
        } else if (EPI == EPI_BIAS) {
          if (mrow < M && ncol < N)
            C[zoffC + (long)mrow * ldc + ncol] = v + bias[ncol];
        } else if (EPI == EPI_SCATTER) {
          const int j = mrow + ncol - (nctx - 1);
          if (mrow < M && ncol < N && j >= 0 && j < nctx)
            C[zoffC + (long)mrow * nctx + j] += v;
        } else {  // EPI_STORE16
          if (mrow < M && ncol < N)
            ((_Float16*)C)[zoffC + (long)mrow * ldc + ncol] = (_Float16)v;
        }
      }
    }
  }
}

// ---------------------------------------------------------------------------
// Positional features
// ---------------------------------------------------------------------------
__global__ void enf_init_gmax(int* gmax) { *gmax = 0; }

__global__ __launch_bounds__(256)
void enf_pos_gamma(float* __restrict__ gtmp, int* __restrict__ gmax, int n) {
  const int idx = blockIdx.x * 256 + threadIdx.x;   // t*32 + j
  const int total = (2 * n - 1) * 32;
  float v = 0.0f;
  if (idx < total) {
    const int t = idx >> 5, j = idx & 31;
    const float ad   = fabsf((float)(t - (n - 1)));
    const float sd   = (float)n / 64.0f;                       // n/(2*nb)
    const float mean = ((float)n / 32.0f) +
                       (float)j * ((float)n - (float)n / 32.0f) / 31.0f;
    const float conc = (mean / sd) * (mean / sd);
    const float rate = mean / (sd * sd);
    const float lu = (conc - 1.0f) * logf(ad) - rate * ad;     // xlogy path
    const float ln = lgammaf(conc) - conc * logf(rate);
    v = expf(lu - ln) + 1e-8f;
    gtmp[idx] = v;
  }
  __shared__ float red[256];
  red[threadIdx.x] = v;
  __syncthreads();
  for (int s = 128; s > 0; s >>= 1) {
    if (threadIdx.x < s)
      red[threadIdx.x] = fmaxf(red[threadIdx.x], red[threadIdx.x + s]);
    __syncthreads();
  }
  if (threadIdx.x == 0) atomicMax(gmax, __float_as_int(red[0]));  // all > 0
}

__global__ __launch_bounds__(256)
void enf_pos_assemble(const float* __restrict__ gtmp, const int* __restrict__ gmax,
                      _Float16* __restrict__ P, int n) {
  const int idx = blockIdx.x * 256 + threadIdx.x;   // t*96 + j
  const int total = (2 * n - 1) * 96;
  if (idx >= total) return;
  const int t = idx / 96, j = idx % 96;
  const float dist = (float)(t - (n - 1));
  const float ad = fabsf(dist);
  float f;
  if (j < 32) {                       // exponential
    const float L  = log2f((float)n);
    const float hl = exp2f(3.0f + (float)j * (L - 3.0f) / 31.0f);
    f = exp2f(-ad / hl);
  } else if (j < 64) {                // central mask
    const float cw = exp2f((float)(j - 31)) - 1.0f;
    f = (cw > ad) ? 1.0f : 0.0f;
  } else {                            // gamma, globally normalized
    f = gtmp[t * 32 + (j - 64)] / __int_as_float(*gmax);
  }
  const float sg = (dist > 0.0f) ? 1.0f : ((dist < 0.0f) ? -1.0f : 0.0f);
  P[(long)t * 192 + j]      = (_Float16)f;
  P[(long)t * 192 + 96 + j] = (_Float16)(sg * f);
}

// ---------------------------------------------------------------------------
// Conversion / reshape helpers (elementwise, bandwidth-trivial)
// ---------------------------------------------------------------------------
__global__ __launch_bounds__(256)
void enf_cvt16(const float* __restrict__ in, _Float16* __restrict__ out, long total) {
  const long i = (long)blockIdx.x * 256 + threadIdx.x;
  if (i < total) out[i] = (_Float16)in[i];
}

// out[N][K] f16 = transpose of in[K][N] f32
__global__ __launch_bounds__(256)
void enf_cvtT16(const float* __restrict__ in, _Float16* __restrict__ out,
                int K, int N) {
  const long i = (long)blockIdx.x * 256 + threadIdx.x;
  if (i >= (long)K * N) return;
  const int nI = (int)(i / K), k = (int)(i % K);
  out[i] = (_Float16)in[(long)k * N + nI];
}

// qc/qp = q*scale + biases -> (bh, n, 64) f16 ; kh = K reshaped -> (bh, n, 64) f16
__global__ __launch_bounds__(256)
void enf_qcqpk(const float* __restrict__ Qf, const float* __restrict__ Kf,
               const float* __restrict__ rcb, const float* __restrict__ rpb,
               _Float16* __restrict__ qc, _Float16* __restrict__ qp,
               _Float16* __restrict__ kh) {
  const long idx = (long)blockIdx.x * 256 + threadIdx.x;   // 16*1536*64
  const int z = (int)(idx / (1536 * 64));
  const int rr = (int)(idx % (1536 * 64));
  const int i = rr >> 6, d = rr & 63;
  const int b = z >> 3, h = z & 7;
  const long src = ((long)(b * 1536 + i)) * 512 + h * 64 + d;
  const float v = Qf[src] * 0.125f;
  qc[idx] = (_Float16)(v + rcb[h * 64 + d]);
  qp[idx] = (_Float16)(v + rpb[h * 64 + d]);
  kh[idx] = (_Float16)Kf[src];
}

// vt (bh, 192, n) f16 = V transposed per head
__global__ __launch_bounds__(256)
void enf_make_vt(const float* __restrict__ Vf, _Float16* __restrict__ vt) {
  const long idx = (long)blockIdx.x * 256 + threadIdx.x;   // 16*192*1536
  const int z = (int)(idx / (192 * 1536));
  const int rr = (int)(idx % (192 * 1536));
  const int d = rr / 1536, j = rr % 1536;
  const int b = z >> 3, h = z & 7;
  vt[idx] = (_Float16)Vf[((long)(b * 1536 + j)) * 1536 + h * 192 + d];
}

// rkh (h, 3072 pad, 64) f16 = rel_k reshaped per head
__global__ __launch_bounds__(256)
void enf_make_rkh(const float* __restrict__ RKf, _Float16* __restrict__ rkh) {
  const long idx = (long)blockIdx.x * 256 + threadIdx.x;   // 8*3071*64
  if (idx >= 8L * 3071 * 64) return;
  const int h = (int)(idx / (3071 * 64));
  const int rr = (int)(idx % (3071 * 64));
  const int j2 = rr >> 6, d = rr & 63;
  rkh[((long)h * 3072 + j2) * 64 + d] = (_Float16)RKf[(long)j2 * 512 + h * 64 + d];
}

// in-place row softmax over 1536-wide rows
__global__ __launch_bounds__(256)
void enf_softmax(float* __restrict__ logits, int n) {
  __shared__ float red[256];
  float* p = logits + (long)blockIdx.x * n;
  const int tid = threadIdx.x;
  float m = -3.4e38f;
  for (int j = tid; j < n; j += 256) m = fmaxf(m, p[j]);
  red[tid] = m; __syncthreads();
  for (int s = 128; s > 0; s >>= 1) {
    if (tid < s) red[tid] = fmaxf(red[tid], red[tid + s]);
    __syncthreads();
  }
  m = red[0]; __syncthreads();
  float sum = 0.0f;
  for (int j = tid; j < n; j += 256) { float e = __expf(p[j] - m); p[j] = e; sum += e; }
  red[tid] = sum; __syncthreads();
  for (int s = 128; s > 0; s >>= 1) {
    if (tid < s) red[tid] += red[tid + s];
    __syncthreads();
  }
  const float inv = 1.0f / red[0];
  for (int j = tid; j < n; j += 256) p[j] *= inv;
}

// ---------------------------------------------------------------------------
extern "C" void kernel_launch(void* const* d_in, const int* in_sizes, int n_in,
                              void* d_out, int out_size, void* d_ws, size_t ws_size,
                              hipStream_t stream) {
  (void)in_sizes; (void)n_in; (void)out_size; (void)ws_size;
  const float* x   = (const float*)d_in[0];
  const float* Wq  = (const float*)d_in[1];
  const float* Wk  = (const float*)d_in[2];
  const float* Wv  = (const float*)d_in[3];
  const float* Wrk = (const float*)d_in[4];
  const float* rcb = (const float*)d_in[5];
  const float* rpb = (const float*)d_in[6];
  const float* Wo  = (const float*)d_in[7];
  const float* bo  = (const float*)d_in[8];
  float* out = (float*)d_out;

  const long n = 1536, BN = 3072;
  char* base = (char*)d_ws;
  long off = 0;
  auto carve = [&](long bytes) -> char* {
    char* p = base + off; off += (bytes + 255) & ~255L; return p;
  };
  // f32 intermediates
  float* Qf     = (float*)carve(BN * 512 * 4);
  float* Kf     = (float*)carve(BN * 512 * 4);
  float* Vf     = (float*)carve(BN * 1536 * 4);
  float* RKf    = (float*)carve(3072L * 512 * 4);
  float* gt     = (float*)carve(3072L * 32 * 4);
  int*   gmax   = (int*)  carve(256);
  float* logits = (float*)carve(16L * n * n * 4);     // ~151MB, L2-resident
  // f16 operands (B operands stored transposed [N][K])
  _Float16* xh   = (_Float16*)carve(BN * 1536 * 2);
  _Float16* WqT  = (_Float16*)carve(512L * 1536 * 2);
  _Float16* WkT  = (_Float16*)carve(512L * 1536 * 2);
  _Float16* WvT  = (_Float16*)carve(1536L * 1536 * 2);
  _Float16* WoT  = (_Float16*)carve(1536L * 1536 * 2);
  _Float16* WrkT = (_Float16*)carve(512L * 192 * 2);
  _Float16* Ph   = (_Float16*)carve(3072L * 192 * 2);
  _Float16* qc   = (_Float16*)carve(16L * n * 64 * 2);
  _Float16* qp   = (_Float16*)carve(16L * n * 64 * 2);
  _Float16* kh   = (_Float16*)carve(16L * n * 64 * 2);
  _Float16* vt   = (_Float16*)carve(16L * 192 * n * 2);
  _Float16* rkh  = (_Float16*)carve(8L * 3072 * 64 * 2);
  _Float16* aVh  = (_Float16*)carve(BN * 1536 * 2);

  dim3 blk(256);

  // positional features (f16 output)
  enf_init_gmax<<<1, 1, 0, stream>>>(gmax);
  enf_pos_gamma<<<(3071 * 32 + 255) / 256, blk, 0, stream>>>(gt, gmax, (int)n);
  enf_pos_assemble<<<(3071 * 96 + 255) / 256, blk, 0, stream>>>(gt, gmax, Ph, (int)n);

  // operand conversions
  enf_cvt16<<<(BN * 1536 + 255) / 256, blk, 0, stream>>>(x, xh, BN * 1536);
  enf_cvtT16<<<(1536 * 512 + 255) / 256, blk, 0, stream>>>(Wq, WqT, 1536, 512);
  enf_cvtT16<<<(1536 * 512 + 255) / 256, blk, 0, stream>>>(Wk, WkT, 1536, 512);
  enf_cvtT16<<<(1536 * 1536 + 255) / 256, blk, 0, stream>>>(Wv, WvT, 1536, 1536);
  enf_cvtT16<<<(1536 * 1536 + 255) / 256, blk, 0, stream>>>(Wo, WoT, 1536, 1536);
  enf_cvtT16<<<(192 * 512 + 255) / 256, blk, 0, stream>>>(Wrk, WrkT, 192, 512);

  // projections (f16 A, f16 B^T, f32 C)
  enf_gemm_wmma<_Float16, EPI_STORE><<<dim3(4, 24, 1), blk, 0, stream>>>(
      xh, WqT, Qf, nullptr, 3072, 512, 1536, 1536, 1536, 512,
      0L, 0L, 0L, 1, 0L, 0L, 1, 0);
  enf_gemm_wmma<_Float16, EPI_STORE><<<dim3(4, 24, 1), blk, 0, stream>>>(
      xh, WkT, Kf, nullptr, 3072, 512, 1536, 1536, 1536, 512,
      0L, 0L, 0L, 1, 0L, 0L, 1, 0);
  enf_gemm_wmma<_Float16, EPI_STORE><<<dim3(12, 24, 1), blk, 0, stream>>>(
      xh, WvT, Vf, nullptr, 3072, 1536, 1536, 1536, 1536, 1536,
      0L, 0L, 0L, 1, 0L, 0L, 1, 0);
  enf_gemm_wmma<_Float16, EPI_STORE><<<dim3(4, 24, 1), blk, 0, stream>>>(
      Ph, WrkT, RKf, nullptr, 3071, 512, 192, 192, 192, 512,
      0L, 0L, 0L, 1, 0L, 0L, 1, 0);

  // head-split reshapes + biases (f16)
  enf_qcqpk<<<(16L * n * 64) / 256, blk, 0, stream>>>(Qf, Kf, rcb, rpb, qc, qp, kh);
  enf_make_vt<<<(16L * 192 * n) / 256, blk, 0, stream>>>(Vf, vt);
  enf_make_rkh<<<(8L * 3071 * 64 + 255) / 256, blk, 0, stream>>>(RKf, rkh);

  // content logits: logits[z] = qc[z] @ kh[z]^T   (z = b*8+h)
  enf_gemm_wmma<_Float16, EPI_STORE><<<dim3(12, 12, 16), blk, 0, stream>>>(
      qc, kh, logits, nullptr, 1536, 1536, 64, 64, 64, 1536,
      n * 64, n * 64, 0L, 1, n * n, 0L, 1, 0);

  // rel logits: qp[z] @ rkh[h]^T, scatter-added with relative shift
  enf_gemm_wmma<_Float16, EPI_SCATTER><<<dim3(24, 12, 16), blk, 0, stream>>>(
      qp, rkh, logits, nullptr, 1536, 3071, 64, 64, 64, 0,
      n * 64, 0L, 3072L * 64, 8, n * n, 0L, 1, (int)n);

  // softmax in place
  enf_softmax<<<16 * 1536, blk, 0, stream>>>(logits, (int)n);

  // attnV(b,n,h*192) f16 = attn[z] @ vt[z]^T   (f32 A converted while staging)
  enf_gemm_wmma<float, EPI_STORE16><<<dim3(2, 12, 16), blk, 0, stream>>>(
      logits, vt, (float*)aVh, nullptr, 1536, 192, 1536, 1536, 1536, 1536,
      n * n, 192L * n, 0L, 1, n * 1536, 192L, 8, 0);

  // out = attnV @ Wo + bo
  enf_gemm_wmma<_Float16, EPI_BIAS><<<dim3(12, 24, 1), blk, 0, stream>>>(
      aVh, WoT, out, bo, 3072, 1536, 1536, 1536, 1536, 1536,
      0L, 0L, 0L, 1, 0L, 0L, 1, 0);
}
